// RelationalNetworkEncoder_48962627174541
// MI455X (gfx1250) — matrix operand order repaired and checked by reading, not compiled, not run on hardware
//
#include <hip/hip_runtime.h>
#include <stdint.h>

typedef __bf16 bf16;
typedef __attribute__((ext_vector_type(16))) __bf16 v16bf;
typedef __attribute__((ext_vector_type(8)))  float  v8f;

typedef __attribute__((ext_vector_type(4))) unsigned tdm_u32x4;
typedef __attribute__((ext_vector_type(8))) int      tdm_i32x8;
typedef __attribute__((ext_vector_type(4))) int      tdm_i32x4;

#define BLK_M 64
#define BLK_N 128
#define BLK_K 32
#define LDAS  40    // BLK_K + 8 pad; 80B row stride (16B aligned) = TDM pad(16 DW interval, 4 DW amount)
#define LDBS  40    // same layout for B: [BLK_N][LDBS], k-contiguous rows

#if defined(__has_builtin)
# if __has_builtin(__builtin_amdgcn_tensor_load_to_lds) && __has_builtin(__builtin_amdgcn_s_wait_tensorcnt)
#  define USE_TDM 1
# endif
#endif
#ifndef USE_TDM
# define USE_TDM 0
#endif

struct Epilogue {
  const float* bn_g;      // if bn_m != null: full BN; else if bn_b != null: bias-only
  const float* bn_b;
  const float* bn_m;
  const float* bn_v;
  const float* residual;  // NCHW residual (out_mode 1) or null
  int relu;
  int out_mode;           // 0: f32 [M][N], 1: f32 NCHW, 2: bf16 [N][M] (transposed, GEMM-B-ready)
  int HW;
  int Cout;
};

union Pack8 { uint4 u; bf16 h[8]; };

__device__ __forceinline__ v8f wmma_bf16(v16bf a, v16bf b, v8f c) {
  return __builtin_amdgcn_wmma_f32_16x16x32_bf16(false, a, false, b, (short)0, c, false, false);
}

// A fragment (16x32): lane L%16 = row M; lanes<16 hold K {0-7,16-23}, lanes>=16 K {8-15,24-31}.
__device__ __forceinline__ v16bf load_a_frag(const bf16* As, int m_tile, int lane) {
  int m  = m_tile + (lane & 15);
  int bk = (lane < 16) ? 0 : 8;
  const bf16* p = As + m * LDAS + bk;
  v16bf a;
  ((uint4*)&a)[0] = *(const uint4*)(p);
  ((uint4*)&a)[1] = *(const uint4*)(p + 16);
  return a;
}

// B fragment (32x16): lane L%16 = column N; lanes<16 hold K {0-7,16-23}, lanes>=16 K {8-15,24-31}.
// Bs is [BLK_N][LDBS] with k contiguous per row -> two b128 loads.
__device__ __forceinline__ v16bf load_b_frag(const bf16* Bs, int n_tile, int lane) {
  int n  = n_tile + (lane & 15);
  int bk = (lane < 16) ? 0 : 8;
  const bf16* p = Bs + n * LDBS + bk;
  v16bf b;
  ((uint4*)&b)[0] = *(const uint4*)(p);
  ((uint4*)&b)[1] = *(const uint4*)(p + 16);
  return b;
}

#if USE_TDM
// One 2D TDM tile load: tile_h rows of tile_w_words DWORDs, global row stride
// stride_words DWORDs; LDS rows padded 16B every 64B (pad codes 3/3) -> 80B stride.
__device__ __forceinline__ void tdm_load_2d(unsigned lds_addr, const void* gptr,
                                            unsigned tile_w_words, unsigned tile_h,
                                            unsigned stride_words)
{
  unsigned long long ga = (unsigned long long)(uintptr_t)gptr;
  tdm_u32x4 g0;
  g0[0] = 1u;                                                   // count=1, user descriptor
  g0[1] = lds_addr;                                             // LDS byte address
  g0[2] = (unsigned)(ga & 0xffffffffu);                         // global_addr[31:0]
  g0[3] = (unsigned)((ga >> 32) & 0x1ffffffu) | 0x80000000u;    // global_addr[56:32] | type=2
  tdm_i32x8 g1;
  g1[0] = (int)((2u << 16) | (1u << 20) | (3u << 22) | (3u << 25)); // data_size=4B, pad_en, 16DW/4DW
  g1[1] = (int)((tile_w_words & 0xffffu) << 16);                // tensor_dim0[15:0]
  g1[2] = (int)((tile_w_words >> 16) | ((tile_h & 0xffffu) << 16)); // dim0 hi | tensor_dim1 lo
  g1[3] = (int)((tile_h >> 16) | ((tile_w_words & 0xffffu) << 16)); // dim1 hi | tile_dim0
  g1[4] = (int)(tile_h & 0xffffu);                              // tile_dim1 (tile_dim2 = 0)
  g1[5] = (int)stride_words;                                    // tensor_dim0_stride[31:0]
  g1[6] = 0;
  g1[7] = 0;
  tdm_i32x4 z4 = {0, 0, 0, 0};
#if defined(__clang_major__) && __clang_major__ >= 23
  tdm_i32x8 z8 = {0, 0, 0, 0, 0, 0, 0, 0};
  __builtin_amdgcn_tensor_load_to_lds(g0, g1, z4, z4, z8, 0);
#else
  __builtin_amdgcn_tensor_load_to_lds(g0, g1, z4, z4, 0);
#endif
}
#endif

__device__ __forceinline__ void epi_tile(v8f acc, int m_base, int n_g, int M, int N,
                                         const Epilogue& ep, void* out) {
  if (n_g >= N) return;
  if (ep.out_mode == 2) {
    // transposed bf16 [N][M]: 8 consecutive m -> one 16B store, directly GEMM-B-consumable
    Pack8 pk;
    #pragma unroll
    for (int v = 0; v < 8; ++v) {
      float val = acc[v];
      if (ep.bn_b) val += ep.bn_b[m_base + v];
      if (ep.relu) val = fmaxf(val, 0.f);
      pk.h[v] = (bf16)val;
    }
    *(uint4*)((bf16*)out + (long)n_g * M + m_base) = pk.u;
    return;
  }
  #pragma unroll
  for (int v = 0; v < 8; ++v) {
    int m = m_base + v;
    float val = acc[v];
    if (ep.bn_m) {
      float s = ep.bn_g[m] * rsqrtf(ep.bn_v[m] + 1e-5f);
      val = val * s + (ep.bn_b[m] - ep.bn_m[m] * s);
    } else if (ep.bn_b) {
      val += ep.bn_b[m];
    }
    long idx;
    if (ep.out_mode == 1) {
      int bi  = n_g / ep.HW;
      int pos = n_g - bi * ep.HW;
      idx = ((long)bi * ep.Cout + m) * ep.HW + pos;
    } else {
      idx = (long)m * N + n_g;
    }
    if (ep.residual) val += ep.residual[idx];
    if (ep.relu) val = fmaxf(val, 0.f);
    ((float*)out)[idx] = val;
  }
}

// C[M,N] = A[M,K](bf16, row-major, lda) * B[N,K](bf16, k-contiguous rows, ldb) + fused epilogue
__global__ __launch_bounds__(256) void wmma_gemm_kernel(
    const bf16* __restrict__ A, int lda,
    const bf16* __restrict__ B, int ldb,
    void* __restrict__ out,
    int M, int N, int K, Epilogue ep)
{
  __shared__ bf16 As[2][BLK_M * LDAS];
  __shared__ bf16 Bs[2][BLK_N * LDBS];

  const int tid  = threadIdx.x;
  const int lane = tid & 31;
  const int wave = tid >> 5;
  const int wave_m  = (wave & 1) * 32;
  const int wave_n  = (wave >> 1) * 32;
  const int block_m = blockIdx.y * BLK_M;
  const int block_n = blockIdx.x * BLK_N;
  const int nk = K / BLK_K;
  const bool interior = (block_n + BLK_N) <= N;   // uniform per workgroup

  v8f acc00 = {}, acc01 = {}, acc10 = {}, acc11 = {};

#if USE_TDM
  if (interior) {
    // ---- TDM-staged path: Tensor Data Mover streams both tiles into padded LDS ----
    const unsigned asAddr[2] = { (unsigned)(uintptr_t)&As[0][0], (unsigned)(uintptr_t)&As[1][0] };
    const unsigned bsAddr[2] = { (unsigned)(uintptr_t)&Bs[0][0], (unsigned)(uintptr_t)&Bs[1][0] };
    if (wave == 0) {
      tdm_load_2d(asAddr[0], A + (long)block_m * lda, BLK_K / 2, BLK_M, (unsigned)(lda / 2));
      tdm_load_2d(bsAddr[0], B + (long)block_n * ldb, BLK_K / 2, BLK_N, (unsigned)(ldb / 2));
    }
    for (int kt = 0; kt < nk; ++kt) {
      if (wave == 0) __builtin_amdgcn_s_wait_tensorcnt(0);  // tile kt landed in LDS
      __syncthreads();                                       // visible to all; buf^1 free
      const int cur = kt & 1;
      if (wave == 0 && kt + 1 < nk) {                        // prefetch next tile via TDM
        const int k0 = (kt + 1) * BLK_K;
        tdm_load_2d(asAddr[cur ^ 1], A + (long)block_m * lda + k0, BLK_K / 2, BLK_M,
                    (unsigned)(lda / 2));
        tdm_load_2d(bsAddr[cur ^ 1], B + (long)block_n * ldb + k0, BLK_K / 2, BLK_N,
                    (unsigned)(ldb / 2));
      }
      const bf16* Ab = As[cur];
      const bf16* Bb = Bs[cur];
      v16bf a0 = load_a_frag(Ab, wave_m + 0,  lane);
      v16bf a1 = load_a_frag(Ab, wave_m + 16, lane);
      v16bf b0 = load_b_frag(Bb, wave_n + 0,  lane);
      v16bf b1 = load_b_frag(Bb, wave_n + 16, lane);
      acc00 = wmma_bf16(a0, b0, acc00);
      acc01 = wmma_bf16(a0, b1, acc01);
      acc10 = wmma_bf16(a1, b0, acc10);
      acc11 = wmma_bf16(a1, b1, acc11);
    }
  } else
#endif
  {
    // ---- register-staged path (edge blocks / fallback): all-b128, per-row guard only ----
    const int ar = tid >> 2;         // A row 0..63
    const int ac = (tid & 3) * 8;    // 0/8/16/24

    uint4 aReg;
    uint4 bReg[2];

    auto load_regs = [&](int k0) {
      aReg = *(const uint4*)(A + (long)(block_m + ar) * lda + k0 + ac);
      #pragma unroll
      for (int h = 0; h < 2; ++h) {
        int chunk = tid + h * 256;           // 512 chunks = 128 rows x 4
        int row = chunk >> 2, col = (chunk & 3) * 8;
        int ng = block_n + row;
        if (interior || ng < N) bReg[h] = *(const uint4*)(B + (long)ng * ldb + k0 + col);
        else                    bReg[h] = make_uint4(0, 0, 0, 0);
      }
    };
    auto store_lds = [&](int buf) {
      *(uint4*)(&As[buf][ar * LDAS + ac]) = aReg;
      #pragma unroll
      for (int h = 0; h < 2; ++h) {
        int chunk = tid + h * 256;
        int row = chunk >> 2, col = (chunk & 3) * 8;
        *(uint4*)(&Bs[buf][row * LDBS + col]) = bReg[h];
      }
    };

    load_regs(0);
    store_lds(0);
    for (int kt = 0; kt < nk; ++kt) {
      const int cur = kt & 1;
      if (kt + 1 < nk) load_regs((kt + 1) * BLK_K);
      __syncthreads();
      const bf16* Ab = As[cur];
      const bf16* Bb = Bs[cur];
      v16bf a0 = load_a_frag(Ab, wave_m + 0,  lane);
      v16bf a1 = load_a_frag(Ab, wave_m + 16, lane);
      v16bf b0 = load_b_frag(Bb, wave_n + 0,  lane);
      v16bf b1 = load_b_frag(Bb, wave_n + 16, lane);
      acc00 = wmma_bf16(a0, b0, acc00);
      acc01 = wmma_bf16(a0, b1, acc01);
      acc10 = wmma_bf16(a1, b0, acc10);
      acc11 = wmma_bf16(a1, b1, acc11);
      if (kt + 1 < nk) store_lds(cur ^ 1);
    }
  }

  const int lhalf = (lane >> 4) * 8;
  const int lcol  = lane & 15;
  epi_tile(acc00, block_m + wave_m + 0  + lhalf, block_n + wave_n + 0  + lcol, M, N, ep, out);
  epi_tile(acc01, block_m + wave_m + 0  + lhalf, block_n + wave_n + 16 + lcol, M, N, ep, out);
  epi_tile(acc10, block_m + wave_m + 16 + lhalf, block_n + wave_n + 0  + lcol, M, N, ep, out);
  epi_tile(acc11, block_m + wave_m + 16 + lhalf, block_n + wave_n + 16 + lcol, M, N, ep, out);
}

// NCHW f32 -> im2col bf16 in GEMM-B layout [Ntot][Kpad] (k contiguous).
// One block column per output position n; threads sweep k -> coalesced writes.
__global__ void im2col_kernel(const float* __restrict__ x, bf16* __restrict__ col,
    int Cin, int H, int W, int KH, int stride, int pad,
    int Ho, int Wo, int Kreal, int Kpad)
{
  int kp = blockIdx.y * blockDim.x + threadIdx.x;
  int n  = blockIdx.x;
  if (kp >= Kpad) return;
  int HoWo = Ho * Wo;
  int b   = n / HoWo;
  int pos = n - b * HoWo;
  int oh  = pos / Wo;
  int ow  = pos - oh * Wo;
  float v = 0.f;
  if (kp < Kreal) {
    int khw = KH * KH;
    int ci = kp / khw;
    int r  = kp - ci * khw;
    int kh = r / KH;
    int kw = r - kh * KH;
    int ih = oh * stride - pad + kh;
    int iw = ow * stride - pad + kw;
    if (ih >= 0 && ih < H && iw >= 0 && iw < W)
      v = x[(((long)b * Cin + ci) * H + ih) * (long)W + iw];
  }
  col[(long)n * Kpad + kp] = (bf16)v;
}

__global__ void wconv_kernel(const float* __restrict__ w, bf16* __restrict__ o,
                             int M, int Kreal, int Kpad)
{
  int i = blockIdx.x * blockDim.x + threadIdx.x;
  if (i >= M * Kpad) return;
  int m = i / Kpad, k = i - m * Kpad;
  o[i] = (bf16)((k < Kreal) ? w[(long)m * Kreal + k] : 0.f);
}

__global__ void maxpool_kernel(const float* __restrict__ x, float* __restrict__ y,
                               int BC, int H, int W, int Ho, int Wo)
{
  int i = blockIdx.x * blockDim.x + threadIdx.x;
  int total = BC * Ho * Wo;
  if (i >= total) return;
  int HW  = Ho * Wo;
  int bc  = i / HW;
  int pos = i - bc * HW;
  int oh = pos / Wo, ow = pos - oh * Wo;
  float m = -3.4e38f;
  #pragma unroll
  for (int kh = 0; kh < 3; ++kh) {
    int ih = oh * 2 - 1 + kh;
    if (ih < 0 || ih >= H) continue;
    #pragma unroll
    for (int kw = 0; kw < 3; ++kw) {
      int iw = ow * 2 - 1 + kw;
      if (iw < 0 || iw >= W) continue;
      m = fmaxf(m, x[((long)bc * H + ih) * W + iw]);
    }
  }
  y[i] = m;
}

// h1[pair n][c] = relu(Ca[c, b*P+q] + Cb[c, b*P+p] + bias[c]); one block per pair,
// threads over c -> coalesced bf16 writes in GEMM-B layout [N2][256].
__global__ void paircombine_kernel(const float* __restrict__ Ca, const float* __restrict__ Cb,
                                   const float* __restrict__ bias, bf16* __restrict__ h1,
                                   int Bn, int P, int C)
{
  long n = blockIdx.x;
  int  c = threadIdx.x;
  int PP = P * P;
  int b = (int)(n / PP);
  int r = (int)(n - (long)b * PP);
  int p = r / P;
  int q = r - p * P;
  long NB = (long)Bn * P;
  float v = Ca[(long)c * NB + b * P + q] + Cb[(long)c * NB + b * P + p] + bias[c];
  h1[n * C + c] = (bf16)fmaxf(v, 0.f);
}

// out[b, c] = sum_j C3[c, b*P + j]
__global__ void reduce_kernel(const float* __restrict__ C3, float* __restrict__ out,
                              int Bn, int P, int C)
{
  int i = blockIdx.x * blockDim.x + threadIdx.x;
  if (i >= Bn * C) return;
  int b = i / C, c = i - b * C;
  const float* p = C3 + (long)c * Bn * P + (long)b * P;
  float s = 0.f;
  for (int j = 0; j < P; ++j) s += p[j];
  out[i] = s;
}

extern "C" void kernel_launch(void* const* d_in, const int* in_sizes, int n_in,
                              void* d_out, int out_size, void* d_ws, size_t ws_size,
                              hipStream_t stream)
{
  (void)in_sizes; (void)n_in; (void)out_size; (void)ws_size;
  auto F = [&](int i) { return (const float*)d_in[i]; };
  const float* im = F(0);

  const size_t MB = 1024ull * 1024ull;
  char*  ws   = (char*)d_ws;
  bf16*  Wbf  = (bf16*)(ws + 0);          //  24 MB: padded bf16 weights
  float* actA = (float*)(ws + 24  * MB);  // 104 MB
  float* actB = (float*)(ws + 128 * MB);  // 104 MB
  bf16*  colB = (bf16*)(ws + 232 * MB);   // 128 MB im2col [N][Kpad]
  float* scB  = (float*)(ws + 360 * MB);  //  14 MB downsample residual
  char*  misc = ws + 374 * MB;            //   8 MB (xin, Ca, Cb)

  // (input-index, M, K_real) for every weight matrix, flatten order of setup_inputs
  struct WEnt { int idx, M, K; };
  const WEnt wl[24] = {
    {1,64,147},
    {6,64,576},{11,64,576},{16,64,576},{21,64,576},
    {26,128,576},{31,128,1152},{36,128,64},{41,128,1152},{46,128,1152},
    {51,256,1152},{56,256,2304},{61,256,128},{66,256,2304},{71,256,2304},
    {76,512,2304},{81,512,4608},{86,512,256},{91,512,4608},{96,512,4608},
    {101,256,512},{103,256,512},{105,256,256},{107,256,256}
  };
  size_t woff[24]; size_t cursor = 0;
  for (int i = 0; i < 24; ++i) {
    int Kp = (wl[i].K + 31) & ~31;
    woff[i] = cursor;
    int total = wl[i].M * Kp;
    wconv_kernel<<<dim3((total + 255) / 256), dim3(256), 0, stream>>>(
        F(wl[i].idx), Wbf + cursor, wl[i].M, wl[i].K, Kp);
    cursor += (size_t)total;
  }

  auto im2col = [&](const float* x, int Cin, int H, int W, int KH, int st, int pad,
                    int Ho, int Wo, int Kreal, int Kpad) {
    int Ntot = 32 * Ho * Wo;
    dim3 g(Ntot, (Kpad + 255) / 256);
    im2col_kernel<<<g, dim3(256), 0, stream>>>(x, colB, Cin, H, W, KH, st, pad, Ho, Wo, Kreal, Kpad);
  };
  auto gemm = [&](const bf16* A, int lda, const bf16* Bm, int ldb, void* out,
                  int M, int N, int K, Epilogue ep) {
    dim3 g((N + BLK_N - 1) / BLK_N, M / BLK_M);
    wmma_gemm_kernel<<<g, dim3(256), 0, stream>>>(A, lda, Bm, ldb, out, M, N, K, ep);
  };
  auto epBN = [&](int bnidx, int relu, const float* resid, int HW, int Cout) {
    Epilogue e{}; e.bn_g = F(bnidx); e.bn_b = F(bnidx + 1);
    e.bn_m = F(bnidx + 2); e.bn_v = F(bnidx + 3);
    e.residual = resid; e.relu = relu; e.out_mode = 1; e.HW = HW; e.Cout = Cout;
    return e;
  };
  auto epBias = [&](int bidx, int relu, int mode) {
    Epilogue e{}; if (bidx >= 0) e.bn_b = F(bidx);
    e.relu = relu; e.out_mode = mode;
    return e;
  };

  // ---- stem: conv7x7 s2 p3 (K 147 -> padded 160) + BN + ReLU, then maxpool 3x3 s2 ----
  im2col(im, 3, 224, 224, 7, 2, 3, 112, 112, 147, 160);
  gemm(Wbf + woff[0], 160, colB, 160, actA, 64, 32 * 112 * 112, 160,
       epBN(2, 1, nullptr, 112 * 112, 64));
  {
    int total = 32 * 64 * 56 * 56;
    maxpool_kernel<<<(total + 255) / 256, 256, 0, stream>>>(actA, actB, 32 * 64, 112, 112, 56, 56);
  }

  float* X = actB; float* Y = actA;   // block input/output always in X, temp in Y
  auto block = [&](int Cin, int Cout, int H, int W, int st,
                   int w1p, int bn1, int w2p, int bn2, int wdp, int bnd) {
    int Ho = H / st, Wo = W / st, HW = Ho * Wo;
    int K1 = Cin * 9;
    im2col(X, Cin, H, W, 3, st, 1, Ho, Wo, K1, K1);
    gemm(Wbf + woff[w1p], K1, colB, K1, Y, Cout, 32 * HW, K1, epBN(bn1, 1, nullptr, HW, Cout));
    const float* resid = X;
    if (wdp >= 0) {
      im2col(X, Cin, H, W, 1, st, 0, Ho, Wo, Cin, Cin);
      gemm(Wbf + woff[wdp], Cin, colB, Cin, scB, Cout, 32 * HW, Cin, epBN(bnd, 0, nullptr, HW, Cout));
      resid = scB;
    }
    int K2 = Cout * 9;
    im2col(Y, Cout, Ho, Wo, 3, 1, 1, Ho, Wo, K2, K2);
    gemm(Wbf + woff[w2p], K2, colB, K2, X, Cout, 32 * HW, K2, epBN(bn2, 1, resid, HW, Cout));
  };

  block(64,  64,  56, 56, 1,  1,  7,  2, 12, -1, -1);
  block(64,  64,  56, 56, 1,  3, 17,  4, 22, -1, -1);
  block(64,  128, 56, 56, 2,  5, 27,  6, 32,  7, 37);
  block(128, 128, 28, 28, 1,  8, 42,  9, 47, -1, -1);
  block(128, 256, 28, 28, 2, 10, 52, 11, 57, 12, 62);
  block(256, 256, 14, 14, 1, 13, 67, 14, 72, -1, -1);
  block(256, 512, 14, 14, 2, 15, 77, 16, 82, 17, 87);
  block(512, 512,  7,  7, 1, 18, 92, 19, 97, -1, -1);

  // ---- projection 512->256 + bias -> xin in GEMM-B layout [1568][256] bf16 ----
  bf16*  xin = (bf16*)misc;               // 1568*256 bf16
  float* Ca  = (float*)(misc + 1 * MB);   // 256*1568 f32
  float* Cb  = (float*)(misc + 3 * MB);
  im2col(X, 512, 7, 7, 1, 1, 0, 7, 7, 512, 512);
  gemm(Wbf + woff[20], 512, colB, 512, xin, 256, 1568, 512, epBias(102, 0, 2));

  // ---- g layer 1 factored: Ca = W1[:, :256]*x_q, Cb = W1[:, 256:]*x_p (f32 [M][N]) ----
  gemm(Wbf + woff[21],       512, xin, 256, Ca, 256, 1568, 256, epBias(-1, 0, 0));
  gemm(Wbf + woff[21] + 256, 512, xin, 256, Cb, 256, 1568, 256, epBias(-1, 0, 0));

  const long N2 = 32L * 2401L;            // 76832 pairs
  bf16* H1 = (bf16*)actA;                 // 39.3 MB bf16 [N2][256]
  paircombine_kernel<<<dim3((unsigned)N2), dim3(256), 0, stream>>>(Ca, Cb, F(104), H1, 32, 49, 256);

  bf16*  H2 = (bf16*)((char*)actA + 48 * MB);   // [N2][256]
  gemm(Wbf + woff[22], 256, H1, 256, H2, 256, (int)N2, 256, epBias(106, 1, 2));
  float* C3 = (float*)colB;               // 78.7 MB f32 [256][N2]
  gemm(Wbf + woff[23], 256, H2, 256, C3, 256, (int)N2, 256, epBias(108, 1, 0));

  reduce_kernel<<<(32 * 256 + 255) / 256, 256, 0, stream>>>(C3, (float*)d_out, 32, 2401, 256);
}
